// Size_TCP_25520695673408
// MI455X (gfx1250) — compile-verified, matrix operand order checked by reading, not run on hardware
//
#include <hip/hip_runtime.h>

// CDNA5 / gfx1250 fused kernel for Size_TCP.
// One workgroup (8 wave32) per batch item; single HBM pass over cloud_top.
// Tile is copied memory->LDS via gfx1250 async-LDS loads (inline asm; the
// clang builtin exists but requires OpenCL-style address-space-qualified
// pointers not expressible in HIP). WMMA fragments are fetched as aligned
// ds_load_b128 pairs from zero-padded f16 staging arrays that alias the
// (dead) tile region.

typedef __attribute__((ext_vector_type(16))) _Float16 v16h;
typedef __attribute__((ext_vector_type(8)))  _Float16 v8h;
typedef __attribute__((ext_vector_type(4)))  _Float16 v4h;
typedef __attribute__((ext_vector_type(2)))  _Float16 v2h;
typedef __attribute__((ext_vector_type(8)))  float    v8f;

#define HH  156
#define HF  78
#define CEN 78
#define ASYNC_TILE 1

union Frag  { v16h v; struct { v8h lo, hi; } p; };
union Bias8 { float4 v[2]; float f[8]; };

#ifdef ASYNC_TILE
__device__ __forceinline__ void async_copy16(const float* g, void* l) {
    // VGLOBAL GLOBAL_LOAD_ASYNC_TO_LDS_B128 (op 98): VDST = LDS byte address,
    // VADDR = 64-bit global address, SADDR = NULL ("off"). ASYNCcnt-tracked.
    // generic LDS pointer low 32 bits == DS-style LDS byte offset.
    const unsigned loff = (unsigned)(uintptr_t)l;
    asm volatile("global_load_async_to_lds_b128 %0, %1, off"
                 :: "v"(loff), "v"(g) : "memory");
}
__device__ __forceinline__ void wait_async0() {
    asm volatile("s_wait_asynccnt 0x0" ::: "memory");
}
#endif

__device__ __forceinline__ float Sv(const float* t, int i, int j) {
    // padded integral-image lookup over in-place 2D prefix-summed tile
    return (i <= 0 || j <= 0) ? 0.0f : t[(i - 1) * HH + (j - 1)];
}

__device__ __forceinline__ float wave_max32(float m) {
#pragma unroll
    for (int off = 16; off > 0; off >>= 1)
        m = fmaxf(m, __shfl_xor(m, off, 32));
    return m;
}

__launch_bounds__(256, 1)
__global__ void size_tcp_fused(const float* __restrict__ cloud_top,
                               const float* __restrict__ pre,
                               const float* __restrict__ w2d,   // (78,156)
                               const float* __restrict__ b2d,   // (78)
                               const float* __restrict__ kw,    // (4,4)
                               const float* __restrict__ qvw,   // (8,4)
                               const float* __restrict__ pw,    // (4,4)
                               const float* __restrict__ pb,    // (4)
                               const float* __restrict__ w1d,   // (64,78)
                               const float* __restrict__ b1d,   // (64)
                               const float* __restrict__ mw1,   // (2,1)
                               const float* __restrict__ mb1,   // (2)
                               const float* __restrict__ mw2,   // (4,2)
                               const float* __restrict__ mb2,   // (4)
                               float* __restrict__ out)         // (b,64,2,2)
{
    extern __shared__ float smem[];
    float* tile  = smem;                  // 24336 f32 (156x156) -- reused below
    float* pool  = tile  + HH * HH;       // 624  (156 x 4): rows 0..77 avg, 78..155 max
    float* cinfo = pool  + HH * 4;        // 320  (80 x 4)
    float* xo    = cinfo + 80 * 4;        // 320  pre-proj attention out
    float* xatt  = xo    + 80 * 4;        // 320  post-proj attention out
    float* gm    = xatt  + 80 * 4;        // 160  (2 x 80) attention logit bases
    float* xout  = gm    + 160;           // 256  (64 x 4) conv1d out
    float* wv    = xout  + 256;           // 4    pre2w gate
    float* b2dl  = wv    + 4;             // 80   conv2d bias (rows 78/79 = 0)
    float* b1dl  = b2dl  + 80;            // 64   conv1d bias

    // f16 staging arrays alias the tile region (dead after integral image):
    _Float16* w2dh  = (_Float16*)tile;        // 80 x 160  (25600 B)
    _Float16* w1dh  = w2dh  + 80 * 160;       // 64 x 96   (12288 B)
    _Float16* poolT = w1dh  + 64 * 96;        // 16 x 160  ( 5120 B)
    _Float16* xattT = poolT + 16 * 160;       // 16 x 96   ( 3072 B)

    const int tid  = threadIdx.x;
    const int lane = tid & 31;
    const int wave = tid >> 5;
    const size_t b = blockIdx.x;

    // small, L2-resident weights: speculative prefetch (global_prefetch_b8)
    __builtin_prefetch(w2d + (tid & 63) * 190, 0, 0);
    __builtin_prefetch(w1d + (tid & 31) * 156, 0, 0);

    // ---------- stage 1: tile -> LDS (gfx1250 async-LDS path) ----------
    {
        const float* src = cloud_top + b * (size_t)(HH * HH);
#ifdef ASYNC_TILE
        for (int i = tid; i < HH * HH / 4; i += 256)
            async_copy16(src + i * 4, tile + i * 4);
        wait_async0();
#else
        const float4* s4 = (const float4*)src;
        float4* d4 = (float4*)tile;
#pragma unroll 4
        for (int i = tid; i < HH * HH / 4; i += 256) d4[i] = s4[i];
#endif
    }
    __syncthreads();

    // ---------- stage 2: max pooling ----------
    // waves 0..3: expanding concentric squares anchored at (78,78), one quadrant each
    if (wave < 4) {
        const int dr = (wave & 2) ? 1 : -1;
        const int dc = (wave & 1) ? 1 : -1;
        float run = tile[CEN * HH + CEN];
        for (int i = 1; i <= 77; ++i) {
            float m = -3.402823466e38f;
            const int r = CEN + dr * i;
            for (int j = lane; j <= i; j += 32)
                m = fmaxf(m, tile[r * HH + CEN + dc * j]);
            const int cl = CEN + dc * i;
            for (int j = lane; j < i; j += 32)
                m = fmaxf(m, tile[(CEN + dr * j) * HH + cl]);
            run = fmaxf(run, wave_max32(m));
            if (lane == 0) pool[(HF + i - 1) * 4 + wave] = run;
        }
    } else {
        // waves 4..7: i==78 clips to the four disjoint 78x78 quadrant blocks
        const int q  = wave - 4;
        const int r0 = (q & 2) ? HF : 0;
        const int c0 = (q & 1) ? HF : 0;
        float m = -3.402823466e38f;
        for (int r = lane; r < HF; r += 32) {               // row-strided, float2 loads
            const float2* row = (const float2*)(tile + (r0 + r) * HH + c0);
#pragma unroll 13
            for (int c = 0; c < HF / 2; ++c) {
                const float2 v = row[c];
                m = fmaxf(m, fmaxf(v.x, v.y));
            }
        }
        m = wave_max32(m);
        if (lane == 0) pool[(HF + 77) * 4 + q] = m;
    }
    __syncthreads();

    // ---------- stage 3: in-place 2D prefix sum (integral image) ----------
    if (tid < HH) {   // row pass: float4 chunks with carried prefix
        float carry = 0.f;
        float4* row = (float4*)(tile + tid * HH);
        for (int k4 = 0; k4 < HH / 4; ++k4) {
            float4 v = row[k4];
            v.x += carry; v.y += v.x; v.z += v.y; v.w += v.z;
            row[k4] = v;
            carry = v.w;
        }
    }
    __syncthreads();
    if (tid < HH) {   // column pass: scalar, conflict-free across threads
        float acc = 0.f;
        for (int k = 0; k < HH; ++k) { acc += tile[k * HH + tid]; tile[k * HH + tid] = acc; }
    }
    __syncthreads();

    // ---------- stage 4: avg pooling from integral image ----------
    if (tid < 77) {
        const int i = tid + 1;
        const int a = CEN - i, p = CEN + 1 + i;
        const float inv = 1.0f / (float)((i + 1) * (i + 1));
        const float scc = Sv(tile, CEN + 1, CEN + 1);
        pool[tid*4+0] = (scc - Sv(tile,a,CEN+1) - Sv(tile,CEN+1,a) + Sv(tile,a,a)) * inv;
        pool[tid*4+1] = (Sv(tile,CEN+1,p) - Sv(tile,a,p) - Sv(tile,CEN+1,CEN) + Sv(tile,a,CEN)) * inv;
        pool[tid*4+2] = (Sv(tile,p,CEN+1) - Sv(tile,CEN,CEN+1) - Sv(tile,p,a) + Sv(tile,CEN,a)) * inv;
        pool[tid*4+3] = (Sv(tile,p,p) - Sv(tile,CEN,p) - Sv(tile,p,CEN) + Sv(tile,CEN,CEN)) * inv;
    } else if (tid == 77) {
        const float inv  = 1.0f / (float)(HF * HF);
        const float sA   = Sv(tile, HF, HF);
        const float sAB  = Sv(tile, HF, HH);
        const float sAC  = Sv(tile, HH, HF);
        const float sAll = Sv(tile, HH, HH);
        pool[77*4+0] = sA * inv;
        pool[77*4+1] = (sAB - sA) * inv;
        pool[77*4+2] = (sAC - sA) * inv;
        pool[77*4+3] = (sAll - sAB - sAC + sA) * inv;
    }
    __syncthreads();   // tile region is now dead -> becomes f16 staging

    // ---------- stage 4b: build zero-padded f16 operands + LDS biases ----------
    {
        // zero all staging (46080 B) with 16B stores
        uint4* z4 = (uint4*)w2dh;
        for (int i = tid; i < (80*160 + 64*96 + 16*160 + 16*96) / 8; i += 256)
            z4[i] = make_uint4(0u, 0u, 0u, 0u);
    }
    __syncthreads();
    // w2d (78x156 f32) -> w2dh (80x160 f16), float4 -> v4h packed stores
    for (int i = tid; i < 78 * 39; i += 256) {
        const int r = i / 39, c4 = (i % 39) * 4;
        const float4 v = *(const float4*)(w2d + r * HH + c4);
        v4h h; h.x = (_Float16)v.x; h.y = (_Float16)v.y;
               h.z = (_Float16)v.z; h.w = (_Float16)v.w;
        *(v4h*)(w2dh + r * 160 + c4) = h;
    }
    // w1d (64x78 f32) -> w1dh (64x96 f16), float2 -> v2h packed stores
    for (int i = tid; i < 64 * 39; i += 256) {
        const int r = i / 39, c2 = (i % 39) * 2;
        const float2 v = *(const float2*)(w1d + r * HF + c2);
        v2h h; h.x = (_Float16)v.x; h.y = (_Float16)v.y;
        *(v2h*)(w1dh + r * 96 + c2) = h;
    }
    // pool (156x4 f32) -> poolT (16x160 f16, transposed, N-padded)
    for (int i = tid; i < HH * 4; i += 256) {
        const int c = i >> 2, n = i & 3;
        poolT[n * 160 + c] = (_Float16)pool[i];
    }
    // biases -> LDS (b2dl padded to 80 rows)
    if (tid < 80) b2dl[tid] = (tid < HF) ? b2d[tid] : 0.f;
    else if (tid >= 128 && tid < 192) b1dl[tid - 128] = b1d[tid - 128];
    __syncthreads();

    // ---------- stage 5: conv2d (78x156)@(156x4) via v_wmma_f32_16x16x32_f16 ----------
    if (wave < 5) {
        const int arow  = wave * 16 + (lane & 15);
        const int khalf = (lane >= 16) ? 8 : 0;
        const int ncol  = lane & 15;
        const _Float16* aptr = w2dh  + arow * 160 + khalf;
        const _Float16* bptr = poolT + ncol * 160 + khalf;
        v8f acc = {};
#pragma unroll
        for (int ks = 0; ks < 5; ++ks) {          // K = 156 padded to 160
            const int kk = ks * 32;
            Frag A, B;
            A.p.lo = *(const v8h*)(aptr + kk);        // K = kk+khalf .. +7
            A.p.hi = *(const v8h*)(aptr + kk + 16);   // K = kk+khalf+16 .. +23
            B.p.lo = *(const v8h*)(bptr + kk);
            B.p.hi = *(const v8h*)(bptr + kk + 16);
            acc = __builtin_amdgcn_wmma_f32_16x16x32_f16(false, A.v, false, B.v,
                                                         (short)0, acc, false, false);
        }
        const int mb = wave * 16 + ((lane >= 16) ? 8 : 0);
        if (ncol < 4) {
            Bias8 bias;                         // two ds_load_b128, one wait
            bias.v[0] = *(const float4*)(b2dl + mb);
            bias.v[1] = *(const float4*)(b2dl + mb + 4);
#pragma unroll
            for (int r = 0; r < 8; ++r)   // rows 78/79 write into padded cinfo, never read
                cinfo[(mb + r) * 4 + ncol] = fmaxf(acc[r] + bias.f[r], 0.f);
        }
    }
    __syncthreads();

    // ---------- stage 6: attention (q,v constant per position -> scalar softmax) ----------
    const float scale = 0.7071067811865476f;  // hd^-0.5, hd=2
    if (tid < 156) {
        const int m = tid % 78, h = tid / 78;
        const float sq0 = qvw[(h*2+0)*4+0] + qvw[(h*2+0)*4+1] + qvw[(h*2+0)*4+2] + qvw[(h*2+0)*4+3];
        const float sq1 = qvw[(h*2+1)*4+0] + qvw[(h*2+1)*4+1] + qvw[(h*2+1)*4+2] + qvw[(h*2+1)*4+3];
        float k0 = 0.f, k1 = 0.f;
#pragma unroll
        for (int ch = 0; ch < 4; ++ch) {
            k0 += cinfo[m*4+ch] * kw[(h*2+0)*4+ch];
            k1 += cinfo[m*4+ch] * kw[(h*2+1)*4+ch];
        }
        gm[h*80+m] = scale * (sq0 * k0 + sq1 * k1);
    }
    __syncthreads();
    if (tid < 156) {
        const int n = tid % 78, h = tid / 78;
        const float vn = (float)n * (1.0f / 77.0f);
        float mx = -3.402823466e38f;
        for (int m = 0; m < 78; ++m) mx = fmaxf(mx, vn * gm[h*80+m]);
        float se = 0.f, sw = 0.f;
        for (int m = 0; m < 78; ++m) {
            const float e = __expf(vn * gm[h*80+m] - mx);
            se += e;
            sw += e * ((float)m * (1.0f / 77.0f));
        }
        const float t = sw / se;
        const float sv0 = qvw[(4+h*2+0)*4+0] + qvw[(4+h*2+0)*4+1] + qvw[(4+h*2+0)*4+2] + qvw[(4+h*2+0)*4+3];
        const float sv1 = qvw[(4+h*2+1)*4+0] + qvw[(4+h*2+1)*4+1] + qvw[(4+h*2+1)*4+2] + qvw[(4+h*2+1)*4+3];
        xo[n*4 + h*2 + 0] = sv0 * t;
        xo[n*4 + h*2 + 1] = sv1 * t;
    }
    __syncthreads();
    if (tid < 312) {   // out projection, (78x4)@(4x4)^T + bias
        const int n = tid >> 2, o = tid & 3;
        float s = pb[o];
#pragma unroll
        for (int ch = 0; ch < 4; ++ch) s += xo[n*4+ch] * pw[o*4+ch];
        xatt[n*4+o] = s;
    }
    __syncthreads();
    // xatt (78x4 f32) -> xattT (16x96 f16, transposed, N-padded)
    for (int i = tid; i < HF * 4; i += 256) {
        const int c = i >> 2, n = i & 3;
        xattT[n * 96 + c] = (_Float16)xatt[i];
    }
    __syncthreads();

    // ---------- stage 7: conv1d (64x78)@(78x4) via v_wmma_f32_16x16x32_f16 ----------
    if (wave < 4) {
        const int arow  = wave * 16 + (lane & 15);
        const int khalf = (lane >= 16) ? 8 : 0;
        const int ncol  = lane & 15;
        const _Float16* aptr = w1dh  + arow * 96 + khalf;
        const _Float16* bptr = xattT + ncol * 96 + khalf;
        v8f acc = {};
#pragma unroll
        for (int ks = 0; ks < 3; ++ks) {          // K = 78 padded to 96
            const int kk = ks * 32;
            Frag A, B;
            A.p.lo = *(const v8h*)(aptr + kk);
            A.p.hi = *(const v8h*)(aptr + kk + 16);
            B.p.lo = *(const v8h*)(bptr + kk);
            B.p.hi = *(const v8h*)(bptr + kk + 16);
            acc = __builtin_amdgcn_wmma_f32_16x16x32_f16(false, A.v, false, B.v,
                                                         (short)0, acc, false, false);
        }
        const int mb = wave * 16 + ((lane >= 16) ? 8 : 0);
        if (ncol < 4) {
            Bias8 bias;
            bias.v[0] = *(const float4*)(b1dl + mb);
            bias.v[1] = *(const float4*)(b1dl + mb + 4);
#pragma unroll
            for (int r = 0; r < 8; ++r)
                xout[(mb + r) * 4 + ncol] = fmaxf(acc[r] + bias.f[r], 0.f);
        }
    }
    // ---------- stage 8: pre2w gate (tiny MLP, one thread) ----------
    if (tid == 0) {
        const float p0 = pre[b];
        const float h0 = tanhf(p0 * mw1[0] + mb1[0]);
        const float h1 = tanhf(p0 * mw1[1] + mb1[1]);
#pragma unroll
        for (int j = 0; j < 4; ++j)
            wv[j] = tanhf(h0 * mw2[j*2+0] + h1 * mw2[j*2+1] + mb2[j]);
    }
    __syncthreads();

    // ---------- stage 9: gated output (b,64,2,2) ----------
    out[b * 256 + tid] = xout[tid] * wv[tid & 3];
}

extern "C" void kernel_launch(void* const* d_in, const int* in_sizes, int n_in,
                              void* d_out, int out_size, void* d_ws, size_t ws_size,
                              hipStream_t stream) {
    (void)n_in; (void)out_size; (void)d_ws; (void)ws_size;
    const int nb = in_sizes[0] / (HH * HH);   // batch
    constexpr int SMEM_FLOATS = HH * HH + HH * 4 + 80 * 4 * 3 + 160 + 256 + 4 + 80 + 64;
    constexpr size_t SMEM = (size_t)SMEM_FLOATS * sizeof(float);  // ~104 KB

    (void)hipFuncSetAttribute(reinterpret_cast<const void*>(size_tcp_fused),
                              hipFuncAttributeMaxDynamicSharedMemorySize, (int)SMEM);

    size_tcp_fused<<<nb, 256, SMEM, stream>>>(
        (const float*)d_in[0],  (const float*)d_in[1],  (const float*)d_in[2],
        (const float*)d_in[3],  (const float*)d_in[4],  (const float*)d_in[5],
        (const float*)d_in[6],  (const float*)d_in[7],  (const float*)d_in[8],
        (const float*)d_in[9],  (const float*)d_in[10], (const float*)d_in[11],
        (const float*)d_in[12], (const float*)d_in[13], (float*)d_out);
}
